// MultiHeadAttentionBatched_36361193128325
// MI455X (gfx1250) — matrix operand order repaired and checked by reading, not compiled, not run on hardware
//
#include <hip/hip_runtime.h>
#include <stdint.h>

// bf16 WMMA fragment types (per CDNA5_HIP.md probe-confirmed signatures)
typedef __attribute__((ext_vector_type(16))) __bf16 v16bf;
typedef __attribute__((ext_vector_type(8)))  float  v8f;
typedef __attribute__((ext_vector_type(4)))  int    v4i;

// pointer-to-addrspace'd-v4i types for the async builtin
typedef __attribute__((address_space(1))) v4i* gv4ip;   // global
typedef __attribute__((address_space(3))) v4i* lv4ip;   // LDS

union Frag16 { uint4 u[2]; v16bf v; unsigned short s[16]; };

// ---- CDNA5 async global->LDS path (ASYNCcnt), guarded for toolchain support
#if defined(__has_builtin)
#if __has_builtin(__builtin_amdgcn_global_load_async_to_lds_b128) && \
    __has_builtin(__builtin_amdgcn_s_wait_asynccnt)
#define HAVE_ASYNC_LDS 1
#endif
#endif
#ifndef HAVE_ASYNC_LDS
#define HAVE_ASYNC_LDS 0
#endif

static __device__ __forceinline__ void async_copy_b128(const void* g, void* l) {
#if HAVE_ASYNC_LDS
  gv4ip gp = (gv4ip)(unsigned long long)(uintptr_t)g;   // flat global addr == AS1 addr
  lv4ip lp = (lv4ip)(unsigned int)(uintptr_t)l;         // low 32 bits == LDS offset
  __builtin_amdgcn_global_load_async_to_lds_b128(gp, lp, 0, 0);
#else
  *(uint4*)l = *(const uint4*)g;
#endif
}

static __device__ __forceinline__ void async_wait() {
#if HAVE_ASYNC_LDS
  __builtin_amdgcn_s_wait_asynccnt(0);
#endif
}

static __device__ __forceinline__ unsigned short f2bf(float f) {
  unsigned int u = __builtin_bit_cast(unsigned int, f);
  u += 0x7FFFu + ((u >> 16) & 1u);          // round-to-nearest-even
  return (unsigned short)(u >> 16);
}

// A-matrix 16x32 bf16 fragment (row per lane, K across VGPRs):
// a[0..7] = K = hi*8 .. hi*8+7 ; a[8..15] = K = 16+hi*8 .. 16+hi*8+7
static __device__ __forceinline__ v16bf ldA(const unsigned short* base, int stride,
                                            int row, int k0, int hi) {
  Frag16 f;
  const unsigned short* p = base + row * stride + k0;
  f.u[0] = *(const uint4*)(p + hi * 8);
  f.u[1] = *(const uint4*)(p + 16 + hi * 8);
  return f.v;
}

// B-matrix 32x16 bf16 fragment from K-contiguous (i.e. transposed) storage:
// lane holds output column "col"; b[0..7] = K = kbase + hi*8 .. +7 ; b[8..15] = K = kbase + 16 + hi*8 .. +7
static __device__ __forceinline__ v16bf ldB(const unsigned short* base, int stride,
                                            int col, int kbase, int hi) {
  Frag16 f;
  const unsigned short* p = base + col * stride + kbase + hi * 8;
  f.u[0] = *(const uint4*)(p);
  f.u[1] = *(const uint4*)(p + 16);
  return f.v;
}

#define WMMA_BF16(a, b, c) \
  __builtin_amdgcn_wmma_f32_16x16x32_bf16(false, (a), false, (b), (short)0, (c), false, false)

// ---------------------------------------------------------------- convert
__global__ __launch_bounds__(256) void cvt_bf16_kernel(const float* __restrict__ in,
                                                       unsigned short* __restrict__ out) {
  int i = (blockIdx.x * 256 + threadIdx.x) * 4;
  float4 f = *(const float4*)(in + i);
  uint2 u;
  u.x = (unsigned int)f2bf(f.x) | ((unsigned int)f2bf(f.y) << 16);
  u.y = (unsigned int)f2bf(f.z) | ((unsigned int)f2bf(f.w) << 16);
  *(uint2*)(out + i) = u;
}

// ---------------------------------------------------------------- QKV GEMM
// C[M=8192, N=3072] = A[8192,1024] * W[1024,3072] + bias ; output scattered
// into bf16 Q/K/V tensors of layout [B,H,T,D].
#define BM 64
#define BN 128
#define BK 32
#define LDT 40   // 32 + 8 pad (keeps 16B alignment: 80B row stride)

__global__ __launch_bounds__(256)
void qkv_gemm_kernel(const unsigned short* __restrict__ A,
                     const unsigned short* __restrict__ W,
                     const float* __restrict__ bias,
                     unsigned short* __restrict__ q16,
                     unsigned short* __restrict__ k16,
                     unsigned short* __restrict__ v16) {
  const int K = 1024, N = 3072, T = 2048, H = 16, D = 64;
  __shared__ __align__(16) unsigned short As[BM * LDT];
  __shared__ __align__(16) unsigned short Bs[BN * LDT];   // transposed: [n][k]

  int m0 = blockIdx.x * BM, n0 = blockIdx.y * BN;
  int tid = threadIdx.x, lane = tid & 31, wave = tid >> 5;
  int wm = wave >> 2, wn = wave & 3;      // 2 x 4 wave grid, 32x32 per wave
  int ln = lane & 15, hi = lane >> 4;

  v8f acc[4];
  v8f zero = {0.f, 0.f, 0.f, 0.f, 0.f, 0.f, 0.f, 0.f};
  acc[0] = zero; acc[1] = zero; acc[2] = zero; acc[3] = zero;

  for (int k0 = 0; k0 < K; k0 += BK) {
    { // A tile: 64x32, 8 bf16 per thread, async DMA straight into LDS
      int idx = tid * 8;
      int row = idx >> 5, col = idx & 31;
      const unsigned short* gp = A + (size_t)(m0 + row) * K + k0 + col;
      if (k0 + BK < K) __builtin_prefetch(gp + BK, 0, 3);
      async_copy_b128(gp, &As[row * LDT + col]);
    }
    { // W tile: 32x128, stored transposed -> K-contiguous B fragments
      int idx = tid * 16;
      int kr = idx >> 7, nc = idx & 127;
      const unsigned short* gp = W + (size_t)(k0 + kr) * N + n0 + nc;
      if (k0 + BK < K) __builtin_prefetch(gp + (size_t)BK * N, 0, 3);
      Frag16 f;
      f.u[0] = *(const uint4*)(gp);
      f.u[1] = *(const uint4*)(gp + 8);
#pragma unroll
      for (int j = 0; j < 16; ++j) Bs[(nc + j) * LDT + kr] = f.s[j];
    }
    async_wait();
    __syncthreads();

    v16bf a0 = ldA(As, LDT, wm * 32 + ln, 0, hi);
    v16bf a1 = ldA(As, LDT, wm * 32 + 16 + ln, 0, hi);
    v16bf b0 = ldB(Bs, LDT, wn * 32 + ln, 0, hi);
    v16bf b1 = ldB(Bs, LDT, wn * 32 + 16 + ln, 0, hi);
    acc[0] = WMMA_BF16(a0, b0, acc[0]);
    acc[1] = WMMA_BF16(a0, b1, acc[1]);
    acc[2] = WMMA_BF16(a1, b0, acc[2]);
    acc[3] = WMMA_BF16(a1, b1, acc[3]);
    __syncthreads();
  }

#pragma unroll
  for (int im = 0; im < 2; ++im)
#pragma unroll
    for (int in = 0; in < 2; ++in) {
      v8f c = acc[im * 2 + in];
      int ncol = n0 + wn * 32 + in * 16 + ln;
      float bv = bias[ncol];
      int which = ncol >> 10, cc = ncol & 1023;
      int h = cc >> 6, d = cc & 63;
      unsigned short* dst = which == 0 ? q16 : (which == 1 ? k16 : v16);
#pragma unroll
      for (int r = 0; r < 8; ++r) {
        int m = m0 + wm * 32 + im * 16 + r + 8 * hi;
        int b = m >> 11, t = m & (T - 1);
        dst[(size_t)((b * H + h) * T + t) * D + d] = f2bf(c[r] + bv);
      }
    }
}

// ---------------------------------------------------------------- attention
// One block = 128 query rows of one (b,h). 8 waves x 16 rows, flash-style
// online softmax over 64-key tiles staged in LDS.
__global__ __launch_bounds__(256)
void attn_kernel(const unsigned short* __restrict__ q16,
                 const unsigned short* __restrict__ k16,
                 const unsigned short* __restrict__ v16,
                 unsigned short* __restrict__ o16) {
  const int T = 2048, D = 64, H = 16, C = 1024;
  const float scale = 0.125f;               // D^-0.5
  __shared__ __align__(16) unsigned short Ks[64 * 72];          // [key][d]
  __shared__ __align__(16) unsigned short Vs[64 * 72];          // [d][key] (transposed)
  __shared__ __align__(16) unsigned short Pb[8 * 16 * 72];      // per-wave P scratch

  int qt = blockIdx.x, bh = blockIdx.y;
  int tid = threadIdx.x, lane = tid & 31, wave = tid >> 5;
  int ln = lane & 15, hi = lane >> 4;
  unsigned short* Pw = Pb + wave * 16 * 72;

  int qrow0 = qt * 128 + wave * 16;
  const unsigned short* qb = q16 + (size_t)bh * T * D;
  const unsigned short* kb = k16 + (size_t)bh * T * D;
  const unsigned short* vb = v16 + (size_t)bh * T * D;

  v16bf aq[2];
#pragma unroll
  for (int dc = 0; dc < 2; ++dc) aq[dc] = ldA(qb, D, qrow0 + ln, dc * 32, hi);

  v8f zero = {0.f, 0.f, 0.f, 0.f, 0.f, 0.f, 0.f, 0.f};
  v8f o[4]; o[0] = zero; o[1] = zero; o[2] = zero; o[3] = zero;
  float mrow[8], lrow[8];
#pragma unroll
  for (int r = 0; r < 8; ++r) { mrow[r] = -1e30f; lrow[r] = 0.f; }

  int nkt = 2 * (qt + 1);                   // causal: keys up to qt*128+127
  for (int kt = 0; kt < nkt; ++kt) {
    int kofs = kt * 64;
    { // stage K via async DMA (no transform), V transposed via registers
      int idx = tid * 16;
      int key = idx >> 6, d = idx & 63;
      const unsigned short* gk = kb + (size_t)(kofs + key) * D + d;
      async_copy_b128(gk,     &Ks[key * 72 + d]);
      async_copy_b128(gk + 8, &Ks[key * 72 + d + 8]);
      const unsigned short* gv = vb + (size_t)(kofs + key) * D + d;
      if (kt + 1 < nkt) {
        __builtin_prefetch(gk + (size_t)64 * D, 0, 3);
        __builtin_prefetch(gv + (size_t)64 * D, 0, 3);
      }
      Frag16 f;
      f.u[0] = *(const uint4*)gv;
      f.u[1] = *(const uint4*)(gv + 8);
#pragma unroll
      for (int j = 0; j < 16; ++j) Vs[(d + j) * 72 + key] = f.s[j];
    }
    async_wait();
    __syncthreads();

    // S = Q K^T  (4 tiles of 16q x 16k, contraction over D=64)
    v8f s[4];
#pragma unroll
    for (int nt = 0; nt < 4; ++nt) {
      s[nt] = zero;
#pragma unroll
      for (int dc = 0; dc < 2; ++dc) {
        v16bf bk = ldB(Ks, 72, nt * 16 + ln, dc * 32, hi);
        s[nt] = WMMA_BF16(aq[dc], bk, s[nt]);
      }
    }
    // scale + causal mask
#pragma unroll
    for (int nt = 0; nt < 4; ++nt)
#pragma unroll
      for (int r = 0; r < 8; ++r) {
        float v = s[nt][r] * scale;
        int kg = kofs + nt * 16 + ln;
        int qg = qrow0 + r + 8 * hi;
        s[nt][r] = (kg <= qg) ? v : -1e30f;
      }
    // online softmax: row stats across 16-lane halves (C-layout rows)
    float corr[8];
#pragma unroll
    for (int r = 0; r < 8; ++r) {
      float mx = fmaxf(fmaxf(s[0][r], s[1][r]), fmaxf(s[2][r], s[3][r]));
      mx = fmaxf(mx, __shfl_xor(mx, 1, 32));
      mx = fmaxf(mx, __shfl_xor(mx, 2, 32));
      mx = fmaxf(mx, __shfl_xor(mx, 4, 32));
      mx = fmaxf(mx, __shfl_xor(mx, 8, 32));
      float nm = fmaxf(mrow[r], mx);
      corr[r] = __expf(mrow[r] - nm);
      mrow[r] = nm;
    }
    float rs[8];
#pragma unroll
    for (int r = 0; r < 8; ++r) rs[r] = 0.f;
#pragma unroll
    for (int nt = 0; nt < 4; ++nt)
#pragma unroll
      for (int r = 0; r < 8; ++r) {
        float p = __expf(s[nt][r] - mrow[r]);
        s[nt][r] = p;
        rs[r] += p;
      }
#pragma unroll
    for (int r = 0; r < 8; ++r) {
      float t = rs[r];
      t += __shfl_xor(t, 1, 32);
      t += __shfl_xor(t, 2, 32);
      t += __shfl_xor(t, 4, 32);
      t += __shfl_xor(t, 8, 32);
      lrow[r] = lrow[r] * corr[r] + t;
    }
#pragma unroll
    for (int dt = 0; dt < 4; ++dt)
#pragma unroll
      for (int r = 0; r < 8; ++r) o[dt][r] *= corr[r];

    // P: C-layout -> per-wave LDS -> A-layout fragments
#pragma unroll
    for (int nt = 0; nt < 4; ++nt)
#pragma unroll
      for (int r = 0; r < 8; ++r)
        Pw[(r + 8 * hi) * 72 + nt * 16 + ln] = f2bf(s[nt][r]);

#pragma unroll
    for (int kc = 0; kc < 2; ++kc) {
      v16bf ap = ldA(Pw, 72, ln, kc * 32, hi);
#pragma unroll
      for (int dt = 0; dt < 4; ++dt) {
        v16bf bv = ldB(Vs, 72, dt * 16 + ln, kc * 32, hi);
        o[dt] = WMMA_BF16(ap, bv, o[dt]);
      }
    }
    __syncthreads();
  }

  int bg = bh >> 4, h = bh & 15;
#pragma unroll
  for (int dt = 0; dt < 4; ++dt)
#pragma unroll
    for (int r = 0; r < 8; ++r) {
      int t = qrow0 + r + 8 * hi;
      int c = h * 64 + dt * 16 + ln;
      o16[(size_t)(bg * T + t) * C + c] = f2bf(o[dt][r] / lrow[r]);
    }
}

// ---------------------------------------------------------------- proj GEMM
__global__ __launch_bounds__(256)
void proj_gemm_kernel(const unsigned short* __restrict__ A,
                      const unsigned short* __restrict__ W,
                      const float* __restrict__ bias,
                      float* __restrict__ out) {
  const int K = 1024, N = 1024;
  __shared__ __align__(16) unsigned short As[BM * LDT];
  __shared__ __align__(16) unsigned short Bs[BN * LDT];

  int m0 = blockIdx.x * BM, n0 = blockIdx.y * BN;
  int tid = threadIdx.x, lane = tid & 31, wave = tid >> 5;
  int wm = wave >> 2, wn = wave & 3;
  int ln = lane & 15, hi = lane >> 4;

  v8f zero = {0.f, 0.f, 0.f, 0.f, 0.f, 0.f, 0.f, 0.f};
  v8f acc[4]; acc[0] = zero; acc[1] = zero; acc[2] = zero; acc[3] = zero;

  for (int k0 = 0; k0 < K; k0 += BK) {
    {
      int idx = tid * 8;
      int row = idx >> 5, col = idx & 31;
      const unsigned short* gp = A + (size_t)(m0 + row) * K + k0 + col;
      if (k0 + BK < K) __builtin_prefetch(gp + BK, 0, 3);
      async_copy_b128(gp, &As[row * LDT + col]);
    }
    {
      int idx = tid * 16;
      int kr = idx >> 7, nc = idx & 127;
      const unsigned short* gp = W + (size_t)(k0 + kr) * N + n0 + nc;
      if (k0 + BK < K) __builtin_prefetch(gp + (size_t)BK * N, 0, 3);
      Frag16 f;
      f.u[0] = *(const uint4*)(gp);
      f.u[1] = *(const uint4*)(gp + 8);
#pragma unroll
      for (int j = 0; j < 16; ++j) Bs[(nc + j) * LDT + kr] = f.s[j];
    }
    async_wait();
    __syncthreads();

    v16bf a0 = ldA(As, LDT, wm * 32 + ln, 0, hi);
    v16bf a1 = ldA(As, LDT, wm * 32 + 16 + ln, 0, hi);
    v16bf b0 = ldB(Bs, LDT, wn * 32 + ln, 0, hi);
    v16bf b1 = ldB(Bs, LDT, wn * 32 + 16 + ln, 0, hi);
    acc[0] = WMMA_BF16(a0, b0, acc[0]);
    acc[1] = WMMA_BF16(a0, b1, acc[1]);
    acc[2] = WMMA_BF16(a1, b0, acc[2]);
    acc[3] = WMMA_BF16(a1, b1, acc[3]);
    __syncthreads();
  }

#pragma unroll
  for (int im = 0; im < 2; ++im)
#pragma unroll
    for (int in = 0; in < 2; ++in) {
      v8f c = acc[im * 2 + in];
      int ncol = n0 + wn * 32 + in * 16 + ln;
      float bv = bias[ncol];
#pragma unroll
      for (int r = 0; r < 8; ++r) {
        int m = m0 + wm * 32 + im * 16 + r + 8 * hi;
        out[(size_t)m * N + ncol] = c[r] + bv;
      }
    }
}

// ---------------------------------------------------------------- launcher
extern "C" void kernel_launch(void* const* d_in, const int* in_sizes, int n_in,
                              void* d_out, int out_size, void* d_ws, size_t ws_size,
                              hipStream_t stream) {
  const int B = 4, T = 2048, C = 1024;
  const float* x      = (const float*)d_in[0];   // [B,T,C]
  const float* w_qkv  = (const float*)d_in[1];   // [C,3C]
  const float* b_qkv  = (const float*)d_in[2];   // [3C]
  const float* w_proj = (const float*)d_in[3];   // [C,C]
  const float* b_proj = (const float*)d_in[4];   // [C]
  float* out = (float*)d_out;                    // [B,T,C]

  const size_t nX  = (size_t)B * T * C;          // 8,388,608
  const size_t nWq = (size_t)C * 3 * C;          // 3,145,728
  const size_t nWp = (size_t)C * C;              // 1,048,576

  unsigned short* ws = (unsigned short*)d_ws;
  unsigned short* x16  = ws;             ws += nX;
  unsigned short* wq16 = ws;             ws += nWq;
  unsigned short* wp16 = ws;             ws += nWp;
  unsigned short* q16  = ws;             ws += nX;
  unsigned short* k16  = ws;             ws += nX;
  unsigned short* v16  = ws;             ws += nX;
  unsigned short* o16  = ws;             ws += nX;

  cvt_bf16_kernel<<<dim3(nX  / 1024), 256, 0, stream>>>(x,      x16);
  cvt_bf16_kernel<<<dim3(nWq / 1024), 256, 0, stream>>>(w_qkv,  wq16);
  cvt_bf16_kernel<<<dim3(nWp / 1024), 256, 0, stream>>>(w_proj, wp16);

  qkv_gemm_kernel<<<dim3(B * T / BM, 3 * C / BN), 256, 0, stream>>>(
      x16, wq16, b_qkv, q16, k16, v16);

  attn_kernel<<<dim3(T / 128, B * 16), 256, 0, stream>>>(q16, k16, v16, o16);

  proj_gemm_kernel<<<dim3(B * T / BM, C / BN), 256, 0, stream>>>(
      o16, wp16, b_proj, out);
}
